// TverskyLayer_20916490731764
// MI455X (gfx1250) — matrix operand order repaired and checked by reading, not compiled, not run on hardware
//
#include <hip/hip_runtime.h>

// Problem constants (from reference): B=65536, D=256, F=512, P=256
#define B_SZ 65536
#define D_SZ 256
#define F_SZ 512
#define P_SZ 256
#define K2_SZ 1024      // concatenated K = 2*F for the fused second GEMM
#define TILE_M 64       // rows of x per workgroup

// LDS layout (bytes); row pads (+8 u16) break 16-lane bank conflicts on A-frag loads
#define SX_STRIDE 264   // u16 elements per row of staged x tile (256 + 8)
#define SC_STRIDE 1032  // u16 elements per row of [xw|xp] tile (1024 + 8)
#define SX_BYTES  (TILE_M * SX_STRIDE * 2)        // 33792
#define SC_BYTES  (TILE_M * SC_STRIDE * 2)        // 132096
#define SWS_OFF   (SX_BYTES + SC_BYTES)           // 165888
#define SPART_OFF (SWS_OFF + TILE_M * 4)          // 166144
#define SMEM_TOTAL (SPART_OFF + 4 * TILE_M * 4)   // 167168 bytes (<320KB WGP LDS)

typedef __attribute__((ext_vector_type(16))) __bf16 bf16x16;
typedef __attribute__((ext_vector_type(8)))  float  floatx8;

union FragCast { uint4 q[2]; bf16x16 v; };

__device__ __forceinline__ unsigned short f2bf(float f) {
  unsigned int u = __builtin_bit_cast(unsigned int, f);
  u += 0x7FFFu + ((u >> 16) & 1u);   // round-to-nearest-even
  return (unsigned short)(u >> 16);
}
__device__ __forceinline__ float bf2f(unsigned short h) {
  unsigned int u = ((unsigned int)h) << 16;
  return __builtin_bit_cast(float, u);
}

__device__ __forceinline__ floatx8 wmma_bf16(uint4 alo, uint4 ahi,
                                             uint4 blo, uint4 bhi, floatx8 c) {
  FragCast a, b;
  a.q[0] = alo; a.q[1] = ahi;
  b.q[0] = blo; b.q[1] = bhi;
  return __builtin_amdgcn_wmma_f32_16x16x32_bf16(
      /*neg_a=*/false, a.v, /*neg_b=*/false, b.v,
      /*c_mod=*/(short)0, c, /*reuse_a=*/false, /*reuse_b=*/false);
}

// ---------------------------------------------------------------------------
// Kernel 0: features (fp32, [F,D]) -> bf16 (row-major, K-contiguous = B-matrix)
// ---------------------------------------------------------------------------
__global__ void cvt_features_kernel(const float* __restrict__ feat,
                                    unsigned short* __restrict__ featBf) {
  int i = blockIdx.x * blockDim.x + threadIdx.x;
  if (i < F_SZ * D_SZ) featBf[i] = f2bf(feat[i]);
}

// ---------------------------------------------------------------------------
// Kernel 1: prototype-side precompute.
//   pf = prototypes @ features.T ; pp = relu(pf) ; pw = pf*pp
//   Bcat[p][f]       = bf16(theta*pw + alpha*pp)      (multiplies xw)
//   Bcat[p][512 + f] = bf16(beta*pw)                  (multiplies xp)
//   pws[p]           = sum_f pw
// 16 blocks x 256 threads; block owns 16 prototypes (no global atomics).
// ---------------------------------------------------------------------------
__global__ void build_proto_kernel(const float* __restrict__ prototypes,
                                   const float* __restrict__ features,
                                   const float* __restrict__ alpha_p,
                                   const float* __restrict__ beta_p,
                                   const float* __restrict__ theta_p,
                                   unsigned short* __restrict__ Bcat,
                                   float* __restrict__ pws) {
  __shared__ float sProto[16][D_SZ];
  __shared__ float sPws[16];
  const int t = threadIdx.x;
  const int pBase = blockIdx.x * 16;

  #pragma unroll
  for (int it = 0; it < 16; ++it) {
    int idx = it * 256 + t;          // 16*256 floats
    sProto[idx >> 8][idx & 255] = prototypes[(pBase + (idx >> 8)) * D_SZ + (idx & 255)];
  }
  if (t < 16) sPws[t] = 0.f;
  __syncthreads();

  const float alpha = alpha_p[0], beta = beta_p[0], theta = theta_p[0];
  float pwPart[16];
  #pragma unroll
  for (int pr = 0; pr < 16; ++pr) pwPart[pr] = 0.f;

  for (int fo = 0; fo < 2; ++fo) {
    const int f = t + fo * 256;
    float acc[16];
    #pragma unroll
    for (int pr = 0; pr < 16; ++pr) acc[pr] = 0.f;
    const float* frow = features + f * D_SZ;
    for (int d = 0; d < D_SZ; d += 4) {
      float4 fv = *(const float4*)(frow + d);
      #pragma unroll
      for (int pr = 0; pr < 16; ++pr) {
        acc[pr] += fv.x * sProto[pr][d]     + fv.y * sProto[pr][d + 1]
                 + fv.z * sProto[pr][d + 2] + fv.w * sProto[pr][d + 3];
      }
    }
    #pragma unroll
    for (int pr = 0; pr < 16; ++pr) {
      float pf = acc[pr];
      float pp = pf > 0.f ? pf : 0.f;
      float pw = pf * pp;
      Bcat[(pBase + pr) * K2_SZ + f]        = f2bf(theta * pw + alpha * pp);
      Bcat[(pBase + pr) * K2_SZ + F_SZ + f] = f2bf(beta * pw);
      pwPart[pr] += pw;
    }
  }
  #pragma unroll
  for (int pr = 0; pr < 16; ++pr) atomicAdd(&sPws[pr], pwPart[pr]);
  __syncthreads();
  if (t < 16) pws[pBase + t] = sPws[t];
}

// ---------------------------------------------------------------------------
// Main fused kernel. One block = 64 rows of x. 256 threads = 8 wave32.
// Stage 1: xf = x_tile @ features.T (WMMA bf16, K=256), relu/square -> LDS.
// Stage 2: out = [xw|xp] @ Bcat.T (WMMA bf16, K=1024) - alpha*rowsum - beta*pws.
// ---------------------------------------------------------------------------
__launch_bounds__(256, 1)
__global__ void tversky_main_kernel(const float* __restrict__ x,
                                    const unsigned short* __restrict__ featBf,
                                    const unsigned short* __restrict__ Bcat,
                                    const float* __restrict__ pws,
                                    const float* __restrict__ alpha_p,
                                    const float* __restrict__ beta_p,
                                    float* __restrict__ out) {
  extern __shared__ char smemRaw[];
  unsigned short* sX    = (unsigned short*)smemRaw;              // [64][264] bf16
  unsigned short* sXcat = (unsigned short*)(smemRaw + SX_BYTES); // [64][1032] bf16
  float* sXws  = (float*)(smemRaw + SWS_OFF);                    // [64]
  float* sPart = (float*)(smemRaw + SPART_OFF);                  // [4][64]

  const int tid    = threadIdx.x;
  const int lane   = tid & 31;
  const int wave   = tid >> 5;
  const int laneLo = lane & 15;
  const int laneHi = lane >> 4;            // 0 or 1
  const int bBase  = blockIdx.x * TILE_M;

  // ---- Stage 0: stage x tile into LDS as bf16 (coalesced b128 loads) ----
  {
    const float4* xv = (const float4*)(x + (size_t)bBase * D_SZ);
    #pragma unroll
    for (int it = 0; it < 16; ++it) {
      int idx = it * 256 + tid;            // 64 rows x 64 float4
      int r = idx >> 6, c4 = idx & 63;
      float4 v = xv[r * 64 + c4];
      unsigned int lo = (unsigned int)f2bf(v.x) | ((unsigned int)f2bf(v.y) << 16);
      unsigned int hi = (unsigned int)f2bf(v.z) | ((unsigned int)f2bf(v.w) << 16);
      *(uint2*)&sX[r * SX_STRIDE + c4 * 4] = make_uint2(lo, hi);
    }
  }
  __syncthreads();

  // ---- Stage 1 GEMM: xf[64,512] ; wave w owns f-columns [w*64, w*64+64) ----
  floatx8 acc1[4][4];
  #pragma unroll
  for (int mt = 0; mt < 4; ++mt)
    #pragma unroll
    for (int nt = 0; nt < 4; ++nt)
      acc1[mt][nt] = (floatx8){0.f,0.f,0.f,0.f,0.f,0.f,0.f,0.f};

  const int nBase1 = wave * 64;
  for (int kk = 0; kk < D_SZ; kk += 32) {
    uint4 aLo[4], aHi[4], bLo[4], bHi[4];
    #pragma unroll
    for (int mt = 0; mt < 4; ++mt) {       // A 16x32: row=lane&15, K chunks k0,k0+16
      const int row = mt * 16 + laneLo;
      const int k0  = kk + (laneHi << 3);
      const unsigned short* p = &sX[row * SX_STRIDE + k0];
      aLo[mt] = *(const uint4*)p;
      aHi[mt] = *(const uint4*)(p + 16);
    }
    #pragma unroll
    for (int nt = 0; nt < 4; ++nt) {       // B 32x16: col=lane&15, K chunks k0,k0+16
      const int f = nBase1 + nt * 16 + laneLo;
      const int d = kk + (laneHi << 3);
      const unsigned short* p = featBf + f * D_SZ + d;
      bLo[nt] = *(const uint4*)p;
      bHi[nt] = *(const uint4*)(p + 16);
    }
    #pragma unroll
    for (int mt = 0; mt < 4; ++mt)
      #pragma unroll
      for (int nt = 0; nt < 4; ++nt)
        acc1[mt][nt] = wmma_bf16(aLo[mt], aHi[mt], bLo[nt], bHi[nt], acc1[mt][nt]);
  }

  // ---- Epilogue 1: relu/square -> sXcat = [xw | xp] in bf16 ----
  #pragma unroll
  for (int mt = 0; mt < 4; ++mt) {
    #pragma unroll
    for (int nt = 0; nt < 4; ++nt) {
      #pragma unroll
      for (int i = 0; i < 8; ++i) {        // D-matrix: M = i + 8*laneHi, N = lane&15
        const int r = mt * 16 + i + (laneHi << 3);
        const int c = nBase1 + nt * 16 + laneLo;
        const float v  = acc1[mt][nt][i];
        const float xp = v > 0.f ? v : 0.f;
        const float xw = v * xp;
        sXcat[r * SC_STRIDE + c]        = f2bf(xw);
        sXcat[r * SC_STRIDE + F_SZ + c] = f2bf(xp);
      }
    }
  }
  __syncthreads();

  // ---- Row sums of xw (for the -alpha * sum(xw) term) ----
  {
    const int r = tid & 63, q = tid >> 6;
    float s = 0.f;
    const unsigned short* rowp = &sXcat[r * SC_STRIDE + q * 128];
    for (int f = 0; f < 128; ++f) s += bf2f(rowp[f]);
    sPart[q * TILE_M + r] = s;
  }
  __syncthreads();
  if (tid < TILE_M)
    sXws[tid] = sPart[tid] + sPart[64 + tid] + sPart[128 + tid] + sPart[192 + tid];
  __syncthreads();

  // ---- Stage 2 GEMM: [64,1024] @ [1024,256] ; wave w owns p-cols [w*32, w*32+32) ----
  floatx8 acc2[4][2];
  #pragma unroll
  for (int mt = 0; mt < 4; ++mt)
    #pragma unroll
    for (int nt = 0; nt < 2; ++nt)
      acc2[mt][nt] = (floatx8){0.f,0.f,0.f,0.f,0.f,0.f,0.f,0.f};

  const int nBase2 = wave * 32;
  for (int kk = 0; kk < K2_SZ; kk += 32) {
    uint4 aLo[4], aHi[4], bLo[2], bHi[2];
    #pragma unroll
    for (int mt = 0; mt < 4; ++mt) {
      const int row = mt * 16 + laneLo;
      const int k0  = kk + (laneHi << 3);
      const unsigned short* p = &sXcat[row * SC_STRIDE + k0];
      aLo[mt] = *(const uint4*)p;
      aHi[mt] = *(const uint4*)(p + 16);
    }
    #pragma unroll
    for (int nt = 0; nt < 2; ++nt) {
      const int pc = nBase2 + nt * 16 + laneLo;
      const int k  = kk + (laneHi << 3);
      const unsigned short* p = Bcat + pc * K2_SZ + k;
      bLo[nt] = *(const uint4*)p;
      bHi[nt] = *(const uint4*)(p + 16);
    }
    #pragma unroll
    for (int mt = 0; mt < 4; ++mt)
      #pragma unroll
      for (int nt = 0; nt < 2; ++nt)
        acc2[mt][nt] = wmma_bf16(aLo[mt], aHi[mt], bLo[nt], bHi[nt], acc2[mt][nt]);
  }

  // ---- Epilogue 2: subtract alpha*rowsum + beta*pws, store fp32 output ----
  const float alpha = alpha_p[0];
  const float beta  = beta_p[0];
  #pragma unroll
  for (int mt = 0; mt < 4; ++mt) {
    #pragma unroll
    for (int nt = 0; nt < 2; ++nt) {
      const int c = nBase2 + nt * 16 + laneLo;
      const float pwsv = pws[c];
      #pragma unroll
      for (int i = 0; i < 8; ++i) {
        const int r = mt * 16 + i + (laneHi << 3);
        const float val = acc2[mt][nt][i] - alpha * sXws[r] - beta * pwsv;
        out[(size_t)(bBase + r) * P_SZ + c] = val;
      }
    }
  }
}

// ---------------------------------------------------------------------------
// Host launcher. Workspace usage: featBf 256KB | Bcat 512KB | pws 1KB (~770KB).
// ---------------------------------------------------------------------------
extern "C" void kernel_launch(void* const* d_in, const int* in_sizes, int n_in,
                              void* d_out, int out_size, void* d_ws, size_t ws_size,
                              hipStream_t stream) {
  (void)in_sizes; (void)n_in; (void)out_size; (void)ws_size;
  const float* x          = (const float*)d_in[0];
  const float* features   = (const float*)d_in[1];
  const float* prototypes = (const float*)d_in[2];
  const float* alpha      = (const float*)d_in[3];
  const float* beta       = (const float*)d_in[4];
  const float* theta      = (const float*)d_in[5];
  float* out = (float*)d_out;

  unsigned short* featBf = (unsigned short*)d_ws;
  unsigned short* Bcat   = (unsigned short*)((char*)d_ws + (size_t)F_SZ * D_SZ * 2);
  float* pws = (float*)((char*)d_ws + (size_t)F_SZ * D_SZ * 2 + (size_t)P_SZ * K2_SZ * 2);

  (void)hipFuncSetAttribute((const void*)tversky_main_kernel,
                            hipFuncAttributeMaxDynamicSharedMemorySize, SMEM_TOTAL);

  cvt_features_kernel<<<(F_SZ * D_SZ) / 256, 256, 0, stream>>>(features, featBf);
  build_proto_kernel<<<P_SZ / 16, 256, 0, stream>>>(prototypes, features,
                                                    alpha, beta, theta, Bcat, pws);
  tversky_main_kernel<<<B_SZ / TILE_M, 256, SMEM_TOTAL, stream>>>(
      x, featBf, Bcat, pws, alpha, beta, out);
}
